// SparseRKANBase_25838523253376
// MI455X (gfx1250) — matrix operand order repaired and checked by reading, not compiled, not run on hardware
//
#include <hip/hip_runtime.h>

#define H 2048
#define BATCH 32
#define TSTEPS 128

typedef __bf16 bf16_t;
typedef __attribute__((ext_vector_type(16))) bf16_t v16bf;
typedef __attribute__((ext_vector_type(8)))  bf16_t v8bf;
typedef __attribute__((ext_vector_type(8)))  float  v8f;

static __device__ __forceinline__ v8f wmma_bf16(v16bf a, v16bf b, v8f c) {
  // (neg_a, A, neg_b, B, c_mod, C, reuse_a, reuse_b)
  return __builtin_amdgcn_wmma_f32_16x16x32_bf16(false, a, false, b, (short)0, c,
                                                 false, false);
}

static __device__ __forceinline__ v16bf cat8(v8bf a, v8bf b) {
  return __builtin_shufflevector(a, b, 0, 1, 2, 3, 4, 5, 6, 7,
                                       8, 9, 10, 11, 12, 13, 14, 15);
}

// ---------------- weight prep: W = w*m split into separate hi/lo bf16 planes -------
__global__ void pack_weights_kernel(const float* __restrict__ w,
                                    const float* __restrict__ m,
                                    bf16_t* __restrict__ hi,
                                    bf16_t* __restrict__ lo, int n) {
  int i = blockIdx.x * blockDim.x + threadIdx.x;
  int stride = gridDim.x * blockDim.x;
  for (; i < n; i += stride) {
    float v = w[i] * m[i];
    bf16_t h = (bf16_t)v;
    hi[i] = h;
    lo[i] = (bf16_t)(v - (float)h);
  }
}

// zero f32 states and the 4 contiguous bf16 activation planes
__global__ void init_state_kernel(float* __restrict__ hf, unsigned* __restrict__ hp) {
  int i = blockIdx.x * blockDim.x + threadIdx.x;  // 65536 threads
  hf[i] = 0.0f;                 // h0f
  hf[BATCH * H + i] = 0.0f;     // h1f
  hp[i] = 0u;                   // 4 bf16 planes = 4*BATCH*H*2 bytes = 2*BATCH*H u32
  hp[BATCH * H + i] = 0u;
}

// ---------------- one-time parallel GEMM: A0 = X @ W0ih^T + b0_ih ------------------
__global__ __launch_bounds__(256) void gemm_a0_kernel(
    const float* __restrict__ X, const bf16_t* __restrict__ Whi,
    const bf16_t* __restrict__ Wlo, const float* __restrict__ bias,
    float* __restrict__ A0) {
  const int wave = threadIdx.x >> 5, lane = threadIdx.x & 31;
  const int tile = blockIdx.x * 8 + wave;       // 32768 tiles
  const int mtile = tile >> 7, ntile = tile & 127;
  const int m0 = mtile * 16, n0 = ntile * 16;
  const int sel = lane >> 4, mA = lane & 15, nB = lane & 15;
  const float*  arow = X   + (size_t)(m0 + mA) * H + sel * 16;
  const bf16_t* wh   = Whi + (size_t)(n0 + nB) * H + sel * 16;
  const bf16_t* wl   = Wlo + (size_t)(n0 + nB) * H + sel * 16;
  v8f acc = {0, 0, 0, 0, 0, 0, 0, 0};
  for (int k0 = 0; k0 < H; k0 += 32) {
    v16bf bh = *(const v16bf*)(wh + k0);
    v16bf bl = *(const v16bf*)(wl + k0);
    float4 a0 = *(const float4*)(arow + k0);
    float4 a1 = *(const float4*)(arow + k0 + 4);
    float4 a2 = *(const float4*)(arow + k0 + 8);
    float4 a3 = *(const float4*)(arow + k0 + 12);
    float af[16] = {a0.x, a0.y, a0.z, a0.w, a1.x, a1.y, a1.z, a1.w,
                    a2.x, a2.y, a2.z, a2.w, a3.x, a3.y, a3.z, a3.w};
    v16bf ah, al;
#pragma unroll
    for (int j = 0; j < 16; ++j) {
      bf16_t hh = (bf16_t)af[j];
      ah[j] = hh;
      al[j] = (bf16_t)(af[j] - (float)hh);
    }
    acc = wmma_bf16(ah, bh, acc);
    acc = wmma_bf16(ah, bl, acc);
    acc = wmma_bf16(al, bh, acc);
  }
  const int n = n0 + nB;
  const float bv = bias[n];
  const int mbase = m0 + 8 * sel;   // D layout: lane l -> N=l&15, M=8*(l>>4)+i
#pragma unroll
  for (int i = 0; i < 8; ++i) A0[(size_t)(mbase + i) * H + n] = acc[i] + bv;
}

// ---------------- per-step GEMM slice over one matrix ------------------------------
static __device__ __forceinline__ void gemm_slice(
    const bf16_t* __restrict__ ahi, const bf16_t* __restrict__ alo,
    const bf16_t* __restrict__ whi, const bf16_t* __restrict__ wlo,
    int n0, int lane, int kbeg, int kend, v8f& acc0, v8f& acc1) {
  const int sel = lane >> 4, mA = lane & 15, nB = lane & 15;
  const bf16_t* wh  = whi + (size_t)(n0 + nB) * H + sel * 16;
  const bf16_t* wl  = wlo + (size_t)(n0 + nB) * H + sel * 16;
  const bf16_t* a0h = ahi + (size_t)mA * H + sel * 16;
  const bf16_t* a0l = alo + (size_t)mA * H + sel * 16;
  const bf16_t* a1h = ahi + (size_t)(mA + 16) * H + sel * 16;
  const bf16_t* a1l = alo + (size_t)(mA + 16) * H + sel * 16;
  for (int k0 = kbeg; k0 < kend; k0 += 32) {
    // weight stream prefetch two iterations (64 k) ahead (global_prefetch_b8)
    __builtin_prefetch((const void*)(wh + k0 + 64), 0, 1);
    __builtin_prefetch((const void*)(wl + k0 + 64), 0, 1);
    v16bf bh  = *(const v16bf*)(wh + k0);
    v16bf bl  = *(const v16bf*)(wl + k0);
    v16bf ah0 = cat8(*(const v8bf*)(a0h + k0), *(const v8bf*)(a0h + k0 + 8));
    v16bf al0 = cat8(*(const v8bf*)(a0l + k0), *(const v8bf*)(a0l + k0 + 8));
    v16bf ah1 = cat8(*(const v8bf*)(a1h + k0), *(const v8bf*)(a1h + k0 + 8));
    v16bf al1 = cat8(*(const v8bf*)(a1l + k0), *(const v8bf*)(a1l + k0 + 8));
    acc0 = wmma_bf16(ah0, bh, acc0);
    acc0 = wmma_bf16(ah0, bl, acc0);
    acc0 = wmma_bf16(al0, bh, acc0);
    acc1 = wmma_bf16(ah1, bh, acc1);
    acc1 = wmma_bf16(ah1, bl, acc1);
    acc1 = wmma_bf16(al1, bh, acc1);
  }
}

// grid 128 (N tiles of 16 cols), block 128 (4 waves split K); produces zpre[32][H]
__global__ __launch_bounds__(128) void step_gemm_kernel(
    const bf16_t* __restrict__ act_hi,    // packed input acts, or nullptr
    const bf16_t* __restrict__ act_lo,
    const bf16_t* __restrict__ Wih_hi, const bf16_t* __restrict__ Wih_lo,
    const bf16_t* __restrict__ h_hi, const bf16_t* __restrict__ h_lo,
    const bf16_t* __restrict__ Whh_hi, const bf16_t* __restrict__ Whh_lo,
    const float* __restrict__ base, int base_stride,  // A0 rows for layer 0, or null
    const float* __restrict__ bias_ih,                // or nullptr (folded into A0)
    const float* __restrict__ bias_hh,
    float* __restrict__ zpre) {
  __shared__ float red[4][2][8][32];
  const int n0 = blockIdx.x * 16;
  const int wave = threadIdx.x >> 5;
  const int lane = threadIdx.x & 31;
  const int kbeg = wave * (H / 4);
  const int kend = kbeg + (H / 4);

  v8f acc0 = {0, 0, 0, 0, 0, 0, 0, 0};
  v8f acc1 = {0, 0, 0, 0, 0, 0, 0, 0};
  if (act_hi) gemm_slice(act_hi, act_lo, Wih_hi, Wih_lo, n0, lane, kbeg, kend,
                         acc0, acc1);
  gemm_slice(h_hi, h_lo, Whh_hi, Whh_lo, n0, lane, kbeg, kend, acc0, acc1);

#pragma unroll
  for (int i = 0; i < 8; ++i) {
    red[wave][0][i][lane] = acc0[i];
    red[wave][1][i][lane] = acc1[i];
  }
  __syncthreads();
  for (int o = threadIdx.x; o < 512; o += 128) {
    const int r = o >> 4, c = o & 15;
    const int tile = r >> 4, mi = r & 15;
    const int i = mi & 7, ln = c + 16 * (mi >> 3);
    float s = red[0][tile][i][ln] + red[1][tile][i][ln]
            + red[2][tile][i][ln] + red[3][tile][i][ln];
    const int n = n0 + c;
    float bv = bias_hh[n];
    if (bias_ih) bv += bias_ih[n];
    if (base)    bv += base[(size_t)r * base_stride + n];
    zpre[(size_t)r * H + n] = s + bv;
  }
}

// ---------------- LayerNorm; writes f32 state + hi/lo bf16 planes (+ out) ----------
__global__ __launch_bounds__(256) void step_ln_kernel(
    const float* __restrict__ zpre, const float* __restrict__ gamma,
    const float* __restrict__ beta, float* __restrict__ hf,
    bf16_t* __restrict__ h_hi, bf16_t* __restrict__ h_lo,
    float* __restrict__ out, int out_stride) {
  __shared__ float s1[256], s2[256];
  const int b = blockIdx.x, tid = threadIdx.x;
  const float* row = zpre + (size_t)b * H;
  float v[8]; float sum = 0.0f, sq = 0.0f;
#pragma unroll
  for (int i = 0; i < 8; ++i) {
    float x = row[i * 256 + tid];
    v[i] = x; sum += x; sq += x * x;
  }
  s1[tid] = sum; s2[tid] = sq; __syncthreads();
  for (int off = 128; off > 0; off >>= 1) {
    if (tid < off) { s1[tid] += s1[tid + off]; s2[tid] += s2[tid + off]; }
    __syncthreads();
  }
  const float mean = s1[0] * (1.0f / H);
  const float var  = s2[0] * (1.0f / H) - mean * mean;
  const float inv  = rsqrtf(var + 1e-5f);
#pragma unroll
  for (int i = 0; i < 8; ++i) {
    const int idx = i * 256 + tid;
    float y = (v[i] - mean) * inv * gamma[idx] + beta[idx];
    hf[(size_t)b * H + idx] = y;
    bf16_t yh = (bf16_t)y;
    h_hi[(size_t)b * H + idx] = yh;
    h_lo[(size_t)b * H + idx] = (bf16_t)(y - (float)yh);
    if (out) out[(size_t)b * out_stride + idx] = y;
  }
}

__global__ void tail_kernel(const float* __restrict__ h0f,
                            const float* __restrict__ h1f,
                            float* __restrict__ dst) {
  int i = blockIdx.x * blockDim.x + threadIdx.x;  // 65536
  dst[i] = h0f[i];
  dst[BATCH * H + i] = h1f[i];
}

extern "C" void kernel_launch(void* const* d_in, const int* in_sizes, int n_in,
                              void* d_out, int out_size, void* d_ws, size_t ws_size,
                              hipStream_t stream) {
  (void)in_sizes; (void)n_in; (void)out_size; (void)ws_size;
  const float* x     = (const float*)d_in[0];
  const float* w_ih  = (const float*)d_in[1];
  const float* b_ih  = (const float*)d_in[2];
  const float* w_hh  = (const float*)d_in[3];
  const float* b_hh  = (const float*)d_in[4];
  const float* m_ih  = (const float*)d_in[5];
  const float* m_hh  = (const float*)d_in[6];
  const float* gamma = (const float*)d_in[7];
  const float* beta  = (const float*)d_in[8];
  float* out = (float*)d_out;

  const size_t HH = (size_t)H * H;       // 4M
  const size_t ST = (size_t)BATCH * H;   // 64K
  char* ws = (char*)d_ws;
  float*  h0f  = (float*)ws;  ws += ST * 4;
  float*  h1f  = (float*)ws;  ws += ST * 4;
  float*  zpre = (float*)ws;  ws += ST * 4;
  bf16_t* h0hi = (bf16_t*)ws; ws += ST * 2;   // 4 contiguous bf16 act planes
  bf16_t* h0lo = (bf16_t*)ws; ws += ST * 2;
  bf16_t* h1hi = (bf16_t*)ws; ws += ST * 2;
  bf16_t* h1lo = (bf16_t*)ws; ws += ST * 2;
  bf16_t* Wih_hi = (bf16_t*)ws; ws += 2 * HH * 2;   // [L][H][H] per plane
  bf16_t* Wih_lo = (bf16_t*)ws; ws += 2 * HH * 2;
  bf16_t* Whh_hi = (bf16_t*)ws; ws += 2 * HH * 2;
  bf16_t* Whh_lo = (bf16_t*)ws; ws += 2 * HH * 2;   // total ws use ~68 MB

  pack_weights_kernel<<<4096, 256, 0, stream>>>(w_ih, m_ih, Wih_hi, Wih_lo,
                                                (int)(2 * HH));
  pack_weights_kernel<<<4096, 256, 0, stream>>>(w_hh, m_hh, Whh_hi, Whh_lo,
                                                (int)(2 * HH));
  init_state_kernel<<<256, 256, 0, stream>>>(h0f, (unsigned*)h0hi);
  // A0 = X @ W0ih^T + b0_ih, staged in d_out[0..B*T*H); consumed before overwrite
  gemm_a0_kernel<<<4096, 256, 0, stream>>>(x, Wih_hi, Wih_lo, b_ih, out);

  for (int t = 0; t < TSTEPS; ++t) {
    float* a0row = out + (size_t)t * H;  // base rows, stride T*H per batch row
    // layer 0: z = A0[t] + h0 @ W0hh^T + b0_hh ; LN -> h0
    step_gemm_kernel<<<128, 128, 0, stream>>>(
        nullptr, nullptr, nullptr, nullptr, h0hi, h0lo, Whh_hi, Whh_lo,
        a0row, TSTEPS * H, nullptr, b_hh, zpre);
    step_ln_kernel<<<32, 256, 0, stream>>>(zpre, gamma, beta, h0f, h0hi, h0lo,
                                           nullptr, 0);
    // layer 1: z = h0 @ W1ih^T + b1_ih + h1 @ W1hh^T + b1_hh ; LN -> h1, out[:,t,:]
    step_gemm_kernel<<<128, 128, 0, stream>>>(
        h0hi, h0lo, Wih_hi + HH, Wih_lo + HH, h1hi, h1lo, Whh_hi + HH, Whh_lo + HH,
        nullptr, 0, b_ih + H, b_hh + H, zpre);
    step_ln_kernel<<<32, 256, 0, stream>>>(zpre, gamma + H, beta + H, h1f, h1hi,
                                           h1lo, out + (size_t)t * H, TSTEPS * H);
  }
  tail_kernel<<<256, 256, 0, stream>>>(h0f, h1f, out + (size_t)BATCH * TSTEPS * H);
}
